// SVGDeterministic_89885075570694
// MI455X (gfx1250) — matrix operand order repaired and checked by reading, not compiled, not run on hardware
//
#include <hip/hip_runtime.h>
#include <hip/hip_bf16.h>
#include <stdint.h>

// ---------------------------------------------------------------------------
// CDNA5 / gfx1250: wave32, WMMA 16x16x32 f16->f32, async global->LDS staging.
// ---------------------------------------------------------------------------

typedef _Float16 half_t;
typedef __attribute__((ext_vector_type(16))) _Float16 v16h;
typedef __attribute__((ext_vector_type(8)))  _Float16 v8h;
typedef __attribute__((ext_vector_type(8)))  float    v8f;

union FragH { v16h v; v8h q[2]; half_t h[16]; };
union FragC { v8f  v; float f[8]; };

__device__ __forceinline__ v8f wmma_f16(v16h a, v16h b, v8f c) {
  // (neg_a, A, neg_b, B, c_mod, C, reuse_a, reuse_b)
  return __builtin_amdgcn_wmma_f32_16x16x32_f16(false, a, false, b, (short)0, c,
                                                false, false);
}

// Async global->LDS copies (ASYNCcnt-tracked DMA path, CDNA5).
__device__ __forceinline__ void async_b128(uint32_t lds_off, const void* g) {
  asm volatile("global_load_async_to_lds_b128 %0, %1, off"
               :: "v"(lds_off), "v"(g) : "memory");
}
__device__ __forceinline__ void async_b64(uint32_t lds_off, const void* g) {
  asm volatile("global_load_async_to_lds_b64 %0, %1, off"
               :: "v"(lds_off), "v"(g) : "memory");
}
// Async loads complete in order -> waiting <=3 retires everything except the
// 3 ops just issued for the next tile.
__device__ __forceinline__ void wait_async_le3() { asm volatile("s_wait_asynccnt 3" ::: "memory"); }
__device__ __forceinline__ void wait_async_le0() { asm volatile("s_wait_asynccnt 0" ::: "memory"); }

__device__ __forceinline__ float dev_sigmoid(float x) { return 1.0f / (1.0f + __expf(-x)); }
__device__ __forceinline__ float dev_gelu(float x)    { return 0.5f * x * (1.0f + erff(x * 0.70710678118654752f)); }
__device__ __forceinline__ float apply_act(float v, int act) {
  if (act == 1) return dev_gelu(v);
  if (act == 2) return tanhf(v);
  return v;
}

// ---------------------------------------------------------------------------
// Packing kernels. B operands are stored fragment-major ("swizzled"):
//   dst[ (k/32)*N + n ]*32 + (k%32)
// so that one lane's 16 B-halves (K = 8*laneHi + h, h<8; 16 + 8*laneHi + h-8, h>=8) are two 16-byte runs,
// and a 32(K)x128(N) tile is one contiguous 8KB block for the async DMA.
// ---------------------------------------------------------------------------

// src: [L][N][K] f32 (torch weight, y = x @ W^T) -> dst swizzled f16.
__global__ void k_pack_linear(const float* __restrict__ src, half_t* __restrict__ dst,
                              int N, int K, int L) {
  long idx = (long)blockIdx.x * blockDim.x + threadIdx.x;
  long per = (long)K * N;
  long tot = per * L;
  if (idx >= tot) return;
  int  l  = (int)(idx / per);
  long p  = idx - (long)l * per;
  int  kr = (int)(p & 31);
  long t2 = p >> 5;
  int  n  = (int)(t2 % N);
  int  kb = (int)(t2 / N);
  int  k  = kb * 32 + kr;
  dst[idx] = (half_t)src[((long)l * N + n) * K + k];
}

// Conv weight pack -> swizzled [Kp/32][Op][32] f16, K index = i*9 + ky*3 + kx.
// flip==0: src OIHW (O,I,3,3). flip==1: src (I,O,3,3), transpose + spatial flip.
__global__ void k_pack_conv(const float* __restrict__ src, half_t* __restrict__ dst,
                            int O, int I, int Kp, int Op, int flip) {
  long idx = (long)blockIdx.x * blockDim.x + threadIdx.x;
  long tot = (long)Kp * Op;
  if (idx >= tot) return;
  int  kr = (int)(idx & 31);
  long t2 = idx >> 5;
  int  o  = (int)(t2 % Op);
  int  kb = (int)(t2 / Op);
  int  k  = kb * 32 + kr;
  float v = 0.0f;
  if (o < O && k < I * 9) {
    int c = k / 9, rr = k - c * 9, ky = rr / 3, kx = rr - ky * 3;
    if (!flip) v = src[(((long)o * I + c) * 3 + ky) * 3 + kx];
    else       v = src[(((long)c * O + o) * 3 + (2 - ky)) * 3 + (2 - kx)];
  }
  dst[idx] = (half_t)v;
}

// Encoder input: f_x_t = concat(x, t, axis=1) -> (1024,1,64,64) f16.
__global__ void k_pack_encin(const float* __restrict__ x, const float* __restrict__ t,
                             half_t* __restrict__ dst) {
  long idx = (long)blockIdx.x * blockDim.x + threadIdx.x;
  const long tot = 1024L * 4096L;
  if (idx >= tot) return;
  int p  = (int)(idx % 4096);
  int n  = (int)(idx / 4096);
  int b  = n >> 5;
  int ch = n & 31;
  float v = (ch < 16) ? x[((long)b * 16 + ch) * 4096 + p]
                      : t[((long)b * 16 + (ch - 16)) * 4096 + p];
  dst[idx] = (half_t)v;
}

// mode 0: z_x rows b*32+j (from z16[1024][256]); mode 1: y_t rows (15+j)*32+b.
__global__ void k_gather(const half_t* __restrict__ src, half_t* __restrict__ dst, int mode) {
  long idx = (long)blockIdx.x * blockDim.x + threadIdx.x;
  const long tot = 512L * 256L;
  if (idx >= tot) return;
  int col = (int)(idx % 256);
  int r   = (int)(idx / 256);
  int b = r >> 4, j = r & 15;
  long srow = mode ? ((long)(15 + j) * 32 + b) : ((long)b * 32 + j);
  dst[idx] = src[srow * 256 + col];
}

// ---------------------------------------------------------------------------
// WMMA GEMM, double-buffered async-LDS staging.
//   C[M,N] = act( A[M,K](f16, row stride strideA) * Bsw[K,N](swizzled f16)
//                 + bias (+ out32 if accum) )
// Block: 256 threads / 8 waves; tile 32(M) x 128(N); each wave: 16x32 (2 WMMA
// sharing one A fragment). Requires M%32==0, N%128==0, K%32==0 (all call sites).
// Every thread issues exactly 3 async ops per K-step (2xB128 + 1xB64).
// ---------------------------------------------------------------------------
__global__ __launch_bounds__(256)
void k_gemm_wmma(const half_t* __restrict__ A, long strideA,
                 const half_t* __restrict__ Bsw,
                 const float* __restrict__ bias,
                 float* __restrict__ out32, half_t* __restrict__ out16,
                 long strideOut, int M, int N, int K, int accum, int act) {
  __shared__ half_t As[2][32 * 32];    // row-major 32(M) x 32(K)
  __shared__ half_t Bs[2][128 * 32];   // fragment-major: [colInTile][32 K]

  const int tid  = threadIdx.x;
  const int lane = tid & 31;
  const int wv   = tid >> 5;
  const int wi   = wv & 1;    // M half
  const int wj   = wv >> 1;   // N quarter (32 cols each)
  const int lhi  = lane >> 4;
  const int ll   = lane & 15;

  const long m0 = (long)blockIdx.x * 32;
  const long n0 = (long)blockIdx.y * 128;

  const uint32_t ldsA[2] = { (uint32_t)(uintptr_t)&As[0][0], (uint32_t)(uintptr_t)&As[1][0] };
  const uint32_t ldsB[2] = { (uint32_t)(uintptr_t)&Bs[0][0], (uint32_t)(uintptr_t)&Bs[1][0] };

  // Per-thread staging coordinates (constant).
  const int arow = tid >> 3, ac8 = tid & 7;       // A: 32 rows x 8 B64 chunks

  auto stage = [&](int k0, int buf) {
    const half_t* gB = Bsw + ((size_t)(k0 >> 5) * N + n0) * 32;   // contiguous 8KB
    async_b128(ldsB[buf] + tid * 16,        gB + tid * 8);
    async_b128(ldsB[buf] + tid * 16 + 4096, gB + tid * 8 + 2048);
    async_b64 (ldsA[buf] + tid * 8,
               A + (m0 + arow) * strideA + k0 + ac8 * 4);
  };

  FragC acc[2];
#pragma unroll
  for (int q = 0; q < 2; ++q)
#pragma unroll
    for (int r = 0; r < 8; ++r) acc[q].f[r] = 0.0f;

  const int nk = K >> 5;
  stage(0, 0);
  for (int kt = 0; kt < nk; ++kt) {
    const int cb = kt & 1;
    __syncthreads();                       // buffer cb^1 no longer being read
    if (kt + 1 < nk) { stage((kt + 1) << 5, cb ^ 1); wait_async_le3(); }
    else             { wait_async_le0(); }
    __syncthreads();                       // tile cb landed for all waves

    FragH a;
    const half_t* ab = &As[cb][(wi * 16 + ll) * 32 + lhi * 8];
    a.q[0] = *(const v8h*)(ab);
    a.q[1] = *(const v8h*)(ab + 16);
#pragma unroll
    for (int q = 0; q < 2; ++q) {
      FragH b;
      const half_t* bb = &Bs[cb][(wj * 32 + q * 16 + ll) * 32 + lhi * 8];
      b.q[0] = *(const v8h*)(bb);
      b.q[1] = *(const v8h*)(bb + 16);
      acc[q].v = wmma_f16(a.v, b.v, acc[q].v);
    }
  }

#pragma unroll
  for (int q = 0; q < 2; ++q) {
    const long n = n0 + wj * 32 + q * 16 + ll;
    const float bs = bias ? bias[n] : 0.0f;
#pragma unroll
    for (int r = 0; r < 8; ++r) {
      const long m = m0 + wi * 16 + lhi * 8 + r;
      float v = acc[q].f[r] + bs;
      if (accum) v += out32[m * strideOut + n];
      v = apply_act(v, act);
      if (out32) out32[m * strideOut + n] = v;
      if (out16) out16[m * strideOut + n] = (half_t)v;
    }
  }
}

// ---------------------------------------------------------------------------
// Implicit-GEMM WMMA conv. Weights swizzled [Kp/32][CoutPad][32] -> the B
// fragment is 2 x global_load_b128 per K-step. A gathered per-lane (f16).
//   mode 0: conv pad1 stride s: iy = oy*s - 1 + ky
//   mode 1: lhs-dilated x2 (tconv, weights pre-flipped): d=oy-1+ky, even, /2
// Wave = 16(pixel) x 16(outC) tile; 8 waves/block; grid.y = CoutPad/16.
// ---------------------------------------------------------------------------
__global__ __launch_bounds__(256)
void k_conv_wmma(const half_t* __restrict__ in, const half_t* __restrict__ wpk,
                 const float* __restrict__ bias,
                 half_t* __restrict__ out16, float* __restrict__ out32,
                 int Nimg, int Cin, int inH, int inW,
                 int Cout, int CoutPad, int outH, int outW,
                 int Kp, int stride, int mode, int act) {
  const int lane = threadIdx.x & 31;
  const int wv   = threadIdx.x >> 5;
  const int lhi  = lane >> 4;
  const int ll   = lane & 15;

  const long M     = (long)Nimg * outH * outW;
  const long mtile = (long)blockIdx.x * 8 + wv;
  if (mtile * 16 >= M) return;                 // wave-uniform
  const int ntile = blockIdx.y;

  const long mA = mtile * 16 + ll;             // A fragment row for this lane
  const int ox = (int)(mA % outW);
  const int oy = (int)((mA / outW) % outH);
  const int nA = (int)(mA / ((long)outW * outH));
  const long inImg = (long)nA * Cin * inH * inW;

  FragC accu;
#pragma unroll
  for (int r = 0; r < 8; ++r) accu.f[r] = 0.0f;

  for (int k0 = 0; k0 < Kp; k0 += 32) {
    if (k0 + 32 < Kp)
      __builtin_prefetch(wpk + ((size_t)((k0 >> 5) + 1) * CoutPad) * 32, 0, 0);

    FragH a, b;
#pragma unroll
    for (int j = 0; j < 16; ++j) {
      const int k = k0 + (j + (j & 8) + (lhi << 3));
      const int c  = k / 9;
      const int rr = k - c * 9;
      const int ky = rr / 3;
      const int kx = rr - ky * 3;
      half_t v = (half_t)0.0f;
      if (c < Cin) {
        int iy, ix;
        bool ok;
        if (mode == 0) {
          iy = oy * stride - 1 + ky;
          ix = ox * stride - 1 + kx;
          ok = (iy >= 0) & (iy < inH) & (ix >= 0) & (ix < inW);
        } else {
          const int dy = oy - 1 + ky;
          const int dx = ox - 1 + kx;
          ok = (dy >= 0) & (dy < 2 * inH) & ((dy & 1) == 0) &
               (dx >= 0) & (dx < 2 * inW) & ((dx & 1) == 0);
          iy = dy >> 1; ix = dx >> 1;
        }
        if (ok) v = in[inImg + ((long)c * inH + iy) * inW + ix];
      }
      a.h[j] = v;
    }
    const half_t* bb = wpk + ((size_t)(k0 >> 5) * CoutPad + ntile * 16 + ll) * 32 + lhi * 8;
    b.q[0] = *(const v8h*)(bb);
    b.q[1] = *(const v8h*)(bb + 16);

    accu.v = wmma_f16(a.v, b.v, accu.v);
  }

  const int oc = ntile * 16 + ll;
  if (oc >= Cout) return;                      // after all WMMAs: wave-safe
  const float bs = bias[oc];
#pragma unroll
  for (int r = 0; r < 8; ++r) {
    const long m = mtile * 16 + lhi * 8 + r;
    const int sox = (int)(m % outW);
    const int soy = (int)((m / outW) % outH);
    const int sn  = (int)(m / ((long)outW * outH));
    float v = apply_act(accu.f[r] + bs, act);
    const long o = (((long)sn * Cout + oc) * outH + soy) * outW + sox;
    if (out16) out16[o] = (half_t)v;
    if (out32) out32[o] = v;
  }
}

// ---------------------------------------------------------------------------
// LSTM pointwise cell: z[b, 0:4H] gates (i,f,g,o); c f32, h out f16.
// ---------------------------------------------------------------------------
__global__ void k_lstm_cell(const float* __restrict__ z, float* __restrict__ c,
                            half_t* __restrict__ h) {
  int idx = blockIdx.x * blockDim.x + threadIdx.x;  // 32*256
  if (idx >= 32 * 256) return;
  int b = idx >> 8, j = idx & 255;
  const float* zr = z + (long)b * 1024;
  float gi = dev_sigmoid(zr[j]);
  float gf = dev_sigmoid(zr[256 + j]);
  float gg = tanhf(zr[512 + j]);
  float go = dev_sigmoid(zr[768 + j]);
  float cn = gf * c[idx] + gi * gg;
  c[idx] = cn;
  h[idx] = (half_t)(go * tanhf(cn));
}

// ---------------------------------------------------------------------------
// MSE: atomic-accumulated mean((p - tg)^2) * inv_n into *out (pre-zeroed).
// ---------------------------------------------------------------------------
__global__ void k_mse(const float* __restrict__ p, const float* __restrict__ tg,
                      long n, float inv_n, float* __restrict__ out) {
  __shared__ float red[256];
  float s = 0.0f;
  for (long i = (long)blockIdx.x * 256 + threadIdx.x; i < n; i += (long)gridDim.x * 256) {
    float d = p[i] - tg[i];
    s += d * d;
  }
  red[threadIdx.x] = s;
  __syncthreads();
  for (int o = 128; o > 0; o >>= 1) {
    if (threadIdx.x < o) red[threadIdx.x] += red[threadIdx.x + o];
    __syncthreads();
  }
  if (threadIdx.x == 0) atomicAdd(out, red[0] * inv_n);
}

// ---------------------------------------------------------------------------
// Host orchestration
// ---------------------------------------------------------------------------
static inline long cdivl(long a, long b) { return (a + b - 1) / b; }

struct DecWeights {
  const half_t *Wdl, *Wdt1, *Wdc1, *Wdt2, *Wdc2, *Wdt3;
  const float *dl_b, *dt1_b, *dc1_b, *dt2_b, *dc2_b, *dt3_b;
  half_t *actA, *actB;
};

static void run_decode(const DecWeights& w, const half_t* zin, float* outF32,
                       hipStream_t stream) {
  k_gemm_wmma<<<dim3(512 / 32, 4096 / 128), 256, 0, stream>>>(
      zin, 256, w.Wdl, w.dl_b, nullptr, w.actA, 4096, 512, 4096, 256, 0, 1);
  k_conv_wmma<<<dim3((unsigned)cdivl(512L * 16 * 16, 128), 4), 256, 0, stream>>>(
      w.actA, w.Wdt1, w.dt1_b, w.actB, nullptr,
      512, 64, 8, 8, 64, 64, 16, 16, 576, 2, 1, 1);
  k_conv_wmma<<<dim3((unsigned)cdivl(512L * 16 * 16, 128), 4), 256, 0, stream>>>(
      w.actB, w.Wdc1, w.dc1_b, w.actA, nullptr,
      512, 64, 16, 16, 64, 64, 16, 16, 576, 1, 0, 1);
  k_conv_wmma<<<dim3((unsigned)cdivl(512L * 32 * 32, 128), 2), 256, 0, stream>>>(
      w.actA, w.Wdt2, w.dt2_b, w.actB, nullptr,
      512, 64, 16, 16, 32, 32, 32, 32, 576, 2, 1, 1);
  k_conv_wmma<<<dim3((unsigned)cdivl(512L * 32 * 32, 128), 2), 256, 0, stream>>>(
      w.actB, w.Wdc2, w.dc2_b, w.actA, nullptr,
      512, 32, 32, 32, 32, 32, 32, 32, 288, 1, 0, 1);
  k_conv_wmma<<<dim3((unsigned)cdivl(512L * 64 * 64, 128), 1), 256, 0, stream>>>(
      w.actA, w.Wdt3, w.dt3_b, nullptr, outF32,
      512, 32, 32, 32, 1, 16, 64, 64, 288, 2, 1, 2);
}

extern "C" void kernel_launch(void* const* d_in, const int* in_sizes, int n_in,
                              void* d_out, int out_size, void* d_ws, size_t ws_size,
                              hipStream_t stream) {
  (void)in_sizes; (void)n_in; (void)out_size; (void)ws_size;
  const float* x     = (const float*)d_in[0];
  const float* t_in  = (const float*)d_in[1];
  const float* ec1_w = (const float*)d_in[2];  const float* ec1_b = (const float*)d_in[3];
  const float* ec2_w = (const float*)d_in[4];  const float* ec2_b = (const float*)d_in[5];
  const float* ec3_w = (const float*)d_in[6];  const float* ec3_b = (const float*)d_in[7];
  const float* ec4_w = (const float*)d_in[8];  const float* ec4_b = (const float*)d_in[9];
  const float* ec5_w = (const float*)d_in[10]; const float* ec5_b = (const float*)d_in[11];
  const float* el_w  = (const float*)d_in[12]; const float* el_b  = (const float*)d_in[13];
  const float* dl_w  = (const float*)d_in[14]; const float* dl_b  = (const float*)d_in[15];
  const float* dt1_w = (const float*)d_in[16]; const float* dt1_b = (const float*)d_in[17];
  const float* dc1_w = (const float*)d_in[18]; const float* dc1_b = (const float*)d_in[19];
  const float* dt2_w = (const float*)d_in[20]; const float* dt2_b = (const float*)d_in[21];
  const float* dc2_w = (const float*)d_in[22]; const float* dc2_b = (const float*)d_in[23];
  const float* dt3_w = (const float*)d_in[24]; const float* dt3_b = (const float*)d_in[25];
  const float* Wih   = (const float*)d_in[26]; const float* Whh   = (const float*)d_in[27];
  const float* bih   = (const float*)d_in[28]; const float* bhh   = (const float*)d_in[29];
  const float* lw    = (const float*)d_in[30]; const float* lb    = (const float*)d_in[31];
  float* out = (float*)d_out;

  // ---- workspace carve (deterministic every call) ----
  char* cur = (char*)d_ws;
  auto alloc = [&cur](size_t bytes) -> char* {
    char* p = cur;
    cur += (bytes + 255) & ~(size_t)255;
    return p;
  };
  auto hAlloc = [&](long elems) { return (half_t*)alloc((size_t)elems * sizeof(half_t)); };
  auto fAlloc = [&](long elems) { return (float*)alloc((size_t)elems * sizeof(float)); };

  half_t* Wec1 = hAlloc(32L * 32);
  half_t* Wec2 = hAlloc(288L * 32);
  half_t* Wec3 = hAlloc(288L * 64);
  half_t* Wec4 = hAlloc(576L * 64);
  half_t* Wec5 = hAlloc(576L * 64);
  half_t* Wel  = hAlloc(4096L * 256);
  half_t* Wdl  = hAlloc(256L * 4096);
  half_t* Wdt1 = hAlloc(576L * 64);
  half_t* Wdc1 = hAlloc(576L * 64);
  half_t* Wdt2 = hAlloc(576L * 32);
  half_t* Wdc2 = hAlloc(288L * 32);
  half_t* Wdt3 = hAlloc(288L * 16);
  half_t* Wihp = hAlloc(10L * 256 * 1024);
  half_t* Whhp = hAlloc(10L * 256 * 1024);
  half_t* Wlw  = hAlloc(256L * 256);
  half_t* encIn = hAlloc(1024L * 4096);
  half_t* actA  = hAlloc(1024L * 32 * 32 * 32);
  half_t* actB  = hAlloc(1024L * 32 * 32 * 32);
  half_t* z16   = hAlloc(1024L * 256);
  half_t* outs  = hAlloc(32L * 32 * 256);
  half_t* hbuf  = hAlloc(10L * 32 * 256);
  half_t* gbuf  = hAlloc(512L * 256);
  float*  zg    = fAlloc(32L * 1024);
  float*  cbuf  = fAlloc(10L * 32 * 256);
  float*  recb  = fAlloc(512L * 4096);

  // ---- pack weights (f16, fragment-major swizzle) ----
  auto packL = [&](const float* s, half_t* d, int N, int K, int L) {
    long tot = (long)L * K * N;
    k_pack_linear<<<(unsigned)cdivl(tot, 256), 256, 0, stream>>>(s, d, N, K, L);
  };
  auto packC = [&](const float* s, half_t* d, int O, int I, int Kp, int Op, int flip) {
    long tot = (long)Kp * Op;
    k_pack_conv<<<(unsigned)cdivl(tot, 256), 256, 0, stream>>>(s, d, O, I, Kp, Op, flip);
  };
  packC(ec1_w, Wec1, 32, 1, 32, 32, 0);
  packC(ec2_w, Wec2, 32, 32, 288, 32, 0);
  packC(ec3_w, Wec3, 64, 32, 288, 64, 0);
  packC(ec4_w, Wec4, 64, 64, 576, 64, 0);
  packC(ec5_w, Wec5, 64, 64, 576, 64, 0);
  packC(dt1_w, Wdt1, 64, 64, 576, 64, 1);
  packC(dc1_w, Wdc1, 64, 64, 576, 64, 0);
  packC(dt2_w, Wdt2, 32, 64, 576, 32, 1);
  packC(dc2_w, Wdc2, 32, 32, 288, 32, 0);
  packC(dt3_w, Wdt3, 1, 32, 288, 16, 1);
  packL(el_w, Wel, 256, 4096, 1);
  packL(dl_w, Wdl, 4096, 256, 1);
  packL(Wih, Wihp, 1024, 256, 10);
  packL(Whh, Whhp, 1024, 256, 10);
  packL(lw, Wlw, 256, 256, 1);

  // ---- encoder ----
  k_pack_encin<<<(unsigned)cdivl(1024L * 4096, 256), 256, 0, stream>>>(x, t_in, encIn);
  k_conv_wmma<<<dim3((unsigned)cdivl(1024L * 32 * 32, 128), 2), 256, 0, stream>>>(
      encIn, Wec1, ec1_b, actA, nullptr, 1024, 1, 64, 64, 32, 32, 32, 32, 32, 2, 0, 1);
  k_conv_wmma<<<dim3((unsigned)cdivl(1024L * 32 * 32, 128), 2), 256, 0, stream>>>(
      actA, Wec2, ec2_b, actB, nullptr, 1024, 32, 32, 32, 32, 32, 32, 32, 288, 1, 0, 1);
  k_conv_wmma<<<dim3((unsigned)cdivl(1024L * 16 * 16, 128), 4), 256, 0, stream>>>(
      actB, Wec3, ec3_b, actA, nullptr, 1024, 32, 32, 32, 64, 64, 16, 16, 288, 2, 0, 1);
  k_conv_wmma<<<dim3((unsigned)cdivl(1024L * 16 * 16, 128), 4), 256, 0, stream>>>(
      actA, Wec4, ec4_b, actB, nullptr, 1024, 64, 16, 16, 64, 64, 16, 16, 576, 1, 0, 1);
  k_conv_wmma<<<dim3((unsigned)cdivl(1024L * 8 * 8, 128), 4), 256, 0, stream>>>(
      actB, Wec5, ec5_b, actA, nullptr, 1024, 64, 16, 16, 64, 64, 8, 8, 576, 2, 0, 1);
  // el: [1024,4096] -> z16 [1024,256]
  k_gemm_wmma<<<dim3(1024 / 32, 256 / 128), 256, 0, stream>>>(
      actA, 4096, Wel, el_b, nullptr, z16, 256, 1024, 256, 4096, 0, 0);

  // ---- init states & loss slots ----
  hipMemsetAsync(out, 0, 2 * sizeof(float), stream);
  hipMemsetAsync(hbuf, 0, 10L * 32 * 256 * sizeof(half_t), stream);
  hipMemsetAsync(cbuf, 0, 10L * 32 * 256 * sizeof(float), stream);

  // ---- LSTM: 32 sequential steps x 10 layers ----
  for (int tt = 0; tt < 32; ++tt) {
    const half_t* xin;
    long sA;
    if (tt < 16) { xin = z16 + (long)tt * 256; sA = 32 * 256; }   // z rows b*32+tt
    else         { xin = outs + (long)(tt - 1) * 32 * 256; sA = 256; }
    for (int l = 0; l < 10; ++l) {
      k_gemm_wmma<<<dim3(1, 1024 / 128), 256, 0, stream>>>(
          xin, sA, Wihp + (long)l * 256 * 1024, bih + (long)l * 1024,
          zg, nullptr, 1024, 32, 1024, 256, 0, 0);
      k_gemm_wmma<<<dim3(1, 1024 / 128), 256, 0, stream>>>(
          hbuf + (long)l * 32 * 256, 256, Whhp + (long)l * 256 * 1024,
          bhh + (long)l * 1024, zg, nullptr, 1024, 32, 1024, 256, 1, 0);
      k_lstm_cell<<<32, 256, 0, stream>>>(zg, cbuf + (long)l * 32 * 256,
                                          hbuf + (long)l * 32 * 256);
      xin = hbuf + (long)l * 32 * 256;
      sA = 256;
    }
    k_gemm_wmma<<<dim3(1, 256 / 128), 256, 0, stream>>>(
        hbuf + 9L * 32 * 256, 256, Wlw, lb, nullptr,
        outs + (long)tt * 32 * 256, 256, 32, 256, 256, 0, 0);
  }

  // ---- decode + losses ----
  DecWeights dw{Wdl, Wdt1, Wdc1, Wdt2, Wdc2, Wdt3,
                dl_b, dt1_b, dc1_b, dt2_b, dc2_b, dt3_b, actA, actB};
  const float inv_n = 1.0f / (512.0f * 4096.0f);

  k_gather<<<512, 256, 0, stream>>>(outs, gbuf, 1);
  run_decode(dw, gbuf, out + 2, stream);
  k_mse<<<512, 256, 0, stream>>>(out + 2, t_in, 512L * 4096, inv_n, out + 0);

  k_gather<<<512, 256, 0, stream>>>(z16, gbuf, 0);
  run_decode(dw, gbuf, recb, stream);
  k_mse<<<512, 256, 0, stream>>>(recb, x, 512L * 4096, inv_n, out + 1);
}
